// ObjectNet_10007273800196
// MI455X (gfx1250) — matrix operand order repaired and checked by reading, not compiled, not run on hardware
//
#include <hip/hip_runtime.h>
#include <hip/hip_bf16.h>

// ---------------------------------------------------------------------------
// PointNet++ (SA-SA-GlobalSA-Head) for MI455X / gfx1250, wave32 + WMMA f16.
// ---------------------------------------------------------------------------

typedef __attribute__((ext_vector_type(16))) _Float16 v16h;
typedef __attribute__((ext_vector_type(8)))  float    v8f;
typedef _Float16 h16;

#define NEG_INF (-3.4e38f)

// ---------------------------------------------------------------------------
// Async global -> LDS copy (CDNA5 VGLOBAL async, tracked by ASYNCcnt).
// Generic pointer low 32 bits == LDS byte offset (ISA 10.2 aperture rule).
// ---------------------------------------------------------------------------
__device__ inline void async_copy_b128(void* lds_dst, const void* gsrc) {
  unsigned int lds_off = (unsigned int)(unsigned long long)lds_dst;
  asm volatile("global_load_async_to_lds_b128 %0, %1, off"
               :: "v"(lds_off), "v"(gsrc) : "memory");
}
__device__ inline void async_wait0() {
  asm volatile("s_wait_asynccnt 0x0" ::: "memory");
}

// ---------------------------------------------------------------------------
// Pack a linear(+BN) layer: W [Cin][Cout] f32 -> Wt f16 [Cout][KPAD] (transposed,
// zero padded); fold BN into per-channel scale/shift:  y = relu?(x@W * s + t)
// ---------------------------------------------------------------------------
__global__ void pack_layer(const float* __restrict__ W, const float* __restrict__ b,
                           const float* __restrict__ gamma, const float* __restrict__ beta,
                           const float* __restrict__ mean,  const float* __restrict__ var,
                           h16* __restrict__ Wt, float* __restrict__ scale,
                           float* __restrict__ shift, int Cin, int Cout, int KPAD) {
  int i = blockIdx.x * blockDim.x + threadIdx.x;
  int total = Cout * KPAD;
  if (i < total) {
    int n = i / KPAD, k = i % KPAD;
    float w = (k < Cin) ? W[(size_t)k * Cout + n] : 0.f;
    Wt[(size_t)n * KPAD + k] = (h16)w;
  }
  if (i < Cout) {
    float s = 1.f, t = b[i];
    if (gamma != nullptr) {
      s = gamma[i] * rsqrtf(var[i] + 1e-5f);
      t = (b[i] - mean[i]) * s + beta[i];
    }
    scale[i] = s;
    shift[i] = t;
  }
}

// ---------------------------------------------------------------------------
// Farthest point sampling: one workgroup per batch, whole cloud in LDS.
// ---------------------------------------------------------------------------
__global__ __launch_bounds__(512)
void fps_kernel(const float* __restrict__ pts, int* __restrict__ idx,
                float* __restrict__ ctrs, int Np, int S) {
  extern __shared__ char smem[];
  float* P    = (float*)smem;           // [Np][3]
  float* mind = P + (size_t)Np * 3;     // [Np]
  float* redv = mind + Np;              // [512]
  int*   redi = (int*)(redv + 512);     // [512]
  float* cur  = (float*)(redi + 512);   // [3]

  const int b = blockIdx.x, t = threadIdx.x, BT = 512;
  const float* pb = pts + (size_t)b * Np * 3;
  for (int i = t; i < Np * 3; i += BT) P[i] = pb[i];
  for (int i = t; i < Np; i += BT) mind[i] = 3.4e38f;
  if (t == 0) {
    idx[(size_t)b * S] = 0;
    cur[0] = P[0]; cur[1] = P[1]; cur[2] = P[2];
    float* cp = ctrs + (size_t)b * S * 3;
    cp[0] = cur[0]; cp[1] = cur[1]; cp[2] = cur[2];
  }
  __syncthreads();

  for (int s = 1; s < S; s++) {
    const float cx = cur[0], cy = cur[1], cz = cur[2];
    float best = -1.f; int bi = 0;
    for (int i = t; i < Np; i += BT) {
      float dx = P[3 * i + 0] - cx;
      float dy = P[3 * i + 1] - cy;
      float dz = P[3 * i + 2] - cz;
      float d2 = dx * dx + dy * dy + dz * dz;
      float m = fminf(mind[i], d2);
      mind[i] = m;
      if (m > best) { best = m; bi = i; }
    }
    redv[t] = best; redi[t] = bi;
    __syncthreads();
    for (int off = 256; off > 0; off >>= 1) {
      if (t < off) {
        if (redv[t + off] > redv[t]) { redv[t] = redv[t + off]; redi[t] = redi[t + off]; }
      }
      __syncthreads();
    }
    if (t == 0) {
      int sel = redi[0];
      idx[(size_t)b * S + s] = sel;
      cur[0] = P[3 * sel + 0]; cur[1] = P[3 * sel + 1]; cur[2] = P[3 * sel + 2];
      float* cp = ctrs + ((size_t)b * S + s) * 3;
      cp[0] = cur[0]; cp[1] = cur[1]; cp[2] = cur[2];
    }
    __syncthreads();
  }
}

// ---------------------------------------------------------------------------
// Radius neighbor query: one thread per center; first K in-radius, pad w/ self.
// ---------------------------------------------------------------------------
__global__ void ball_query(const float* __restrict__ pts, const float* __restrict__ ctr,
                           const int* __restrict__ fidx, int* __restrict__ nbr,
                           float r2, int Np, int S, int B) {
  int g = blockIdx.x * blockDim.x + threadIdx.x;
  if (g >= B * S) return;
  int b = g / S;
  const float* pb = pts + (size_t)b * Np * 3;
  const float* c = ctr + (size_t)g * 3;
  const float cx = c[0], cy = c[1], cz = c[2];
  int* out = nbr + (size_t)g * 64;
  int cnt = 0;
  for (int p = 0; p < Np && cnt < 64; p++) {
    float dx = pb[3 * p + 0] - cx;
    float dy = pb[3 * p + 1] - cy;
    float dz = pb[3 * p + 2] - cz;
    if (dx * dx + dy * dy + dz * dz <= r2) out[cnt++] = p;
  }
  int self = fidx[g];
  for (; cnt < 64; cnt++) out[cnt] = self;
}

// ---------------------------------------------------------------------------
// WMMA fragment loads (wave32; A and B both in contiguous k = 16*(lane/16) + s order).
// ---------------------------------------------------------------------------
__device__ inline v16h ld_afrag(const h16* __restrict__ pa) {
  v16h a;
#pragma unroll
  for (int i = 0; i < 8; i++) { a[i] = pa[i]; a[i + 8] = pa[8 + i]; }
  return a;
}
__device__ inline v16h ld_bfrag(const h16* __restrict__ pb) {
  v16h b;
#pragma unroll
  for (int i = 0; i < 16; i++) b[i] = pb[i];
  return b;
}

// 16x64 output group per wave: 4 accumulators, A fragment loaded once per
// k-chunk and reused by 4 independent WMMAs (hides WMMA->WMMA latency, cuts
// A traffic 4x vs per-tile reload).
__device__ inline void wmma_group4(const h16* __restrict__ A, int lda,
                                   const h16* __restrict__ Wt, int ldw,
                                   int rowbase, int nbase, int kchunks, v8f c[4]) {
  const int lane = threadIdx.x & 31;
  const h16* pa0 = A + (size_t)(rowbase + (lane & 15)) * lda + ((lane >> 4) << 4);
  const h16* pb0 = Wt + (size_t)(nbase + (lane & 15)) * ldw + ((lane >> 4) << 4);
  for (int k = 0; k < kchunks; k++) {
    v16h a = ld_afrag(pa0 + k * 32);
#pragma unroll
    for (int i = 0; i < 4; i++) {
      v16h b = ld_bfrag(pb0 + (size_t)i * 16 * ldw + k * 32);
      c[i] = __builtin_amdgcn_wmma_f32_16x16x32_f16(false, a, false, b, (short)0,
                                                    c[i], false, false);
    }
  }
}

// ---------------------------------------------------------------------------
// Fused set-abstraction stage: 128 edges (2 centers x K=64) per workgroup.
// gather -> L1(WMMA,BN,ReLU) -> L2 -> L3(linear) -> max over K -> xout f16.
// 8 waves, each owning a private 16-row M-subtile across all layers.
// ---------------------------------------------------------------------------
template <int CIN_PAD, int CX, int C1, int C2, int C3, bool HAS_X>
__global__ __launch_bounds__(256)
void sa_fused(const float* __restrict__ pts, const float* __restrict__ ctr,
              const int* __restrict__ nbr, const h16* __restrict__ xin,
              const h16* __restrict__ W1, const float* __restrict__ s1, const float* __restrict__ t1,
              const h16* __restrict__ W2, const float* __restrict__ s2, const float* __restrict__ t2,
              const h16* __restrict__ W3, const float* __restrict__ s3, const float* __restrict__ t3,
              h16* __restrict__ xout, int Np, int S) {
  extern __shared__ char smem[];
  h16* A0  = (h16*)smem;                 // [128][CIN_PAD]
  h16* A1  = A0 + 128 * CIN_PAD;         // [128][C1]
  h16* A2  = A1 + 128 * C1;              // [128][C2]
  h16* LW1 = A2 + 128 * C2;              // [C1][CIN_PAD]
  h16* LW2 = LW1 + C1 * CIN_PAD;         // [C2][C1]
  h16* LW3 = LW2 + C2 * C1;              // [C3][C2]
  float* tmpm = (float*)(LW3 + C3 * C2); // [8][C3]

  const int b = blockIdx.x, c0 = blockIdx.y * 2, t = threadIdx.x;

  // Async-stage all three weight tiles into LDS (overlaps with gather below).
  for (int i = t; i < C1 * CIN_PAD / 8; i += 256) async_copy_b128(LW1 + i * 8, W1 + i * 8);
  for (int i = t; i < C2 * C1 / 8; i += 256)      async_copy_b128(LW2 + i * 8, W2 + i * 8);
  for (int i = t; i < C3 * C2 / 8; i += 256)      async_copy_b128(LW3 + i * 8, W3 + i * 8);

  if (t < 128) {
    int sc = c0 + (t >> 6);
    int j  = t & 63;
    int p  = nbr[((size_t)b * S + sc) * 64 + j];
    const float* pp = pts + ((size_t)b * Np + p) * 3;
    const float* cc = ctr + ((size_t)b * S + sc) * 3;
    h16* row = A0 + (size_t)t * CIN_PAD;
    int o = 0;
    if (HAS_X) {
      const h16* xr = xin + ((size_t)b * Np + p) * CX;
      for (int i = 0; i < CX; i++) row[o++] = xr[i];
    }
    row[o++] = (h16)(pp[0] - cc[0]);
    row[o++] = (h16)(pp[1] - cc[1]);
    row[o++] = (h16)(pp[2] - cc[2]);
    for (; o < CIN_PAD; o++) row[o] = (h16)0.f;
  }
  async_wait0();
  __syncthreads();

  const int wave = t >> 5, lane = t & 31;
  const int rowbase = wave * 16;
  const int rofs = (lane >> 4) << 3;
  const v8f zero = {};

  // Layer 1: A0 [128][CIN_PAD] -> A1 [128][C1]
  for (int nb = 0; nb < C1; nb += 64) {
    v8f c[4] = {zero, zero, zero, zero};
    wmma_group4(A0, CIN_PAD, LW1, CIN_PAD, rowbase, nb, CIN_PAD / 32, c);
#pragma unroll
    for (int i = 0; i < 4; i++) {
      int col = nb + i * 16 + (lane & 15);
      float sc = s1[col], sh = t1[col];
#pragma unroll
      for (int j = 0; j < 8; j++) {
        float v = fmaxf(c[i][j] * sc + sh, 0.f);
        A1[(size_t)(rowbase + rofs + j) * C1 + col] = (h16)v;
      }
    }
  }
  // Layer 2: A1 -> A2
  for (int nb = 0; nb < C2; nb += 64) {
    v8f c[4] = {zero, zero, zero, zero};
    wmma_group4(A1, C1, LW2, C1, rowbase, nb, C1 / 32, c);
#pragma unroll
    for (int i = 0; i < 4; i++) {
      int col = nb + i * 16 + (lane & 15);
      float sc = s2[col], sh = t2[col];
#pragma unroll
      for (int j = 0; j < 8; j++) {
        float v = fmaxf(c[i][j] * sc + sh, 0.f);
        A2[(size_t)(rowbase + rofs + j) * C2 + col] = (h16)v;
      }
    }
  }
  // Layer 3 (plain linear) + per-wave max over its 16 rows
  for (int nb = 0; nb < C3; nb += 64) {
    v8f c[4] = {zero, zero, zero, zero};
    wmma_group4(A2, C2, LW3, C2, rowbase, nb, C2 / 32, c);
#pragma unroll
    for (int i = 0; i < 4; i++) {
      int col = nb + i * 16 + (lane & 15);
      float sc = s3[col], sh = t3[col];
      float m = NEG_INF;
#pragma unroll
      for (int j = 0; j < 8; j++) m = fmaxf(m, c[i][j] * sc + sh);
      m = fmaxf(m, __shfl_xor(m, 16));
      if (lane < 16) tmpm[wave * C3 + col] = m;
    }
  }
  __syncthreads();

  // Cross-wave max (waves 0-3 = center 0, waves 4-7 = center 1) -> global f16
  for (int i = t; i < 2 * C3; i += 256) {
    int cc = i / C3, ch = i % C3;
    float m = tmpm[(cc * 4 + 0) * C3 + ch];
    m = fmaxf(m, tmpm[(cc * 4 + 1) * C3 + ch]);
    m = fmaxf(m, tmpm[(cc * 4 + 2) * C3 + ch]);
    m = fmaxf(m, tmpm[(cc * 4 + 3) * C3 + ch]);
    xout[((size_t)b * S + c0 + cc) * C3 + ch] = (h16)m;
  }
}

// ---------------------------------------------------------------------------
// Build SA3 input rows: concat(x2[256] f16, p2[3]) zero-padded to 288.
// ---------------------------------------------------------------------------
__global__ void build_a3(const h16* __restrict__ x2, const float* __restrict__ p2,
                         h16* __restrict__ A3, int rows) {
  int i = blockIdx.x * blockDim.x + threadIdx.x;
  if (i >= rows * 288) return;
  int r = i / 288, c = i % 288;
  h16 v;
  if (c < 256)      v = x2[(size_t)r * 256 + c];
  else if (c < 259) v = (h16)p2[(size_t)r * 3 + (c - 256)];
  else              v = (h16)0.f;
  A3[i] = v;
}

// ---------------------------------------------------------------------------
// Streaming WMMA GEMM: out = act(A[M][KPAD] @ Wt[N][KPAD]^T * s + t).
// 8 waves / block, 128x64 tile per block, k-outer with 4 accumulators.
// ---------------------------------------------------------------------------
__global__ __launch_bounds__(256)
void gemm_wmma(const h16* __restrict__ A, int KPAD, const h16* __restrict__ Wt,
               const float* __restrict__ scale, const float* __restrict__ shift,
               int relu, h16* __restrict__ outH, float* __restrict__ outF,
               int M, int N) {
  const int wave = threadIdx.x >> 5, lane = threadIdx.x & 31;
  const int row0 = blockIdx.x * 128 + wave * 16;
  const int colblk = blockIdx.y * 64;
  const int rofs = (lane >> 4) << 3;
  const h16* pa0 = A + (size_t)(row0 + (lane & 15)) * KPAD + ((lane >> 4) << 4);
  const h16* pb0 = Wt + (size_t)(colblk + (lane & 15)) * KPAD + ((lane >> 4) << 4);

  const v8f zero = {};
  v8f c[4] = {zero, zero, zero, zero};
  for (int k0 = 0; k0 < KPAD; k0 += 32) {
    __builtin_prefetch(pa0 + k0 + 64, 0, 0);
    v16h a = ld_afrag(pa0 + k0);
#pragma unroll
    for (int i = 0; i < 4; i++) {
      v16h b = ld_bfrag(pb0 + (size_t)i * 16 * KPAD + k0);
      c[i] = __builtin_amdgcn_wmma_f32_16x16x32_f16(false, a, false, b, (short)0,
                                                    c[i], false, false);
    }
  }
#pragma unroll
  for (int i = 0; i < 4; i++) {
    int col = colblk + i * 16 + (lane & 15);
    float sc = scale[col], sh = shift[col];
#pragma unroll
    for (int j = 0; j < 8; j++) {
      float v = c[i][j] * sc + sh;
      if (relu) v = fmaxf(v, 0.f);
      int row = row0 + rofs + j;
      if (outH) outH[(size_t)row * N + col] = (h16)v;
      else      outF[(size_t)row * N + col] = v;
    }
  }
}

// ---------------------------------------------------------------------------
// Global max pool over S2 points per batch.
// ---------------------------------------------------------------------------
__global__ void maxpool(const float* __restrict__ act, float* __restrict__ g,
                        int S, int C) {
  int i = blockIdx.x * blockDim.x + threadIdx.x;
  if (i >= 8 * C) return;
  int b = i / C, ch = i % C;
  const float* p = act + (size_t)b * S * C + ch;
  float m = NEG_INF;
  for (int s = 0; s < S; s++) m = fmaxf(m, p[(size_t)s * C]);
  g[i] = m;
}

// ---------------------------------------------------------------------------
// Head MLP (M=8, pure VALU) + log_softmax; one block per batch row.
// ---------------------------------------------------------------------------
__global__ __launch_bounds__(256)
void head_kernel(const float* __restrict__ g,
                 const h16* __restrict__ W0, const float* __restrict__ b0,
                 const h16* __restrict__ W1, const float* __restrict__ b1,
                 const h16* __restrict__ W2, const float* __restrict__ b2,
                 float* __restrict__ out) {
  __shared__ float gb[1024], h1[512], h2[256], lg[8];
  const int b = blockIdx.x, t = threadIdx.x;
  for (int i = t; i < 1024; i += 256) gb[i] = g[(size_t)b * 1024 + i];
  __syncthreads();
  for (int n = t; n < 512; n += 256) {
    const h16* w = W0 + (size_t)n * 1024;
    float acc = 0.f;
    for (int k = 0; k < 1024; k++) acc += gb[k] * (float)w[k];
    h1[n] = fmaxf(acc + b0[n], 0.f);
  }
  __syncthreads();
  if (t < 256) {
    const h16* w = W1 + (size_t)t * 512;
    float acc = 0.f;
    for (int k = 0; k < 512; k++) acc += h1[k] * (float)w[k];
    h2[t] = fmaxf(acc + b1[t], 0.f);
  }
  __syncthreads();
  if (t < 8) {
    const h16* w = W2 + (size_t)t * 256;
    float acc = 0.f;
    for (int k = 0; k < 256; k++) acc += h2[k] * (float)w[k];
    lg[t] = acc + b2[t];
  }
  __syncthreads();
  if (t == 0) {
    float mx = lg[0];
    for (int i = 1; i < 8; i++) mx = fmaxf(mx, lg[i]);
    float se = 0.f;
    for (int i = 0; i < 8; i++) se += expf(lg[i] - mx);
    float lse = mx + logf(se);
    for (int i = 0; i < 8; i++) out[b * 8 + i] = lg[i] - lse;
  }
}

// ---------------------------------------------------------------------------
// Host side
// ---------------------------------------------------------------------------
namespace {
struct LP {
  const float *W, *b, *gamma, *beta, *mean, *var;
  int Cin, Cout, KPAD, relu;
};
}

extern "C" void kernel_launch(void* const* d_in, const int* in_sizes, int n_in,
                              void* d_out, int out_size, void* d_ws, size_t ws_size,
                              hipStream_t stream) {
  const int B = 8, N = 4096, S1 = 2048, S2 = 512;

  // --- resolve input layout (dict insertion order vs jax sorted-key order) ---
  const bool posFirst = (in_sizes[0] == B * N * 3);
  const float* pos = (const float*)(posFirst ? d_in[0] : d_in[n_in - 1]);

  LP L[12];  // 0-2 sa1, 3-5 sa2, 6-8 sa3, 9-11 head
  const int sa1d[4] = {3, 64, 64, 128};
  const int sa2d[4] = {131, 128, 128, 256};
  const int sa3d[4] = {259, 256, 512, 1024};
  const int hdd[4]  = {1024, 512, 256, 8};
  const int kpads[12] = {32, 64, 64, 160, 128, 128, 288, 256, 512, 1024, 512, 256};

  int cur = posFirst ? 1 : 0;
  auto fill = [&](int li0, const int* dims, bool norm) {
    for (int i = 0; i < 3; i++) {
      LP& p = L[li0 + i];
      bool bn = norm && (i < 2);
      p.W = (const float*)d_in[cur++];
      p.b = (const float*)d_in[cur++];
      if (bn) {
        if (posFirst) { p.gamma = (const float*)d_in[cur++]; p.beta = (const float*)d_in[cur++]; }
        else          { p.beta  = (const float*)d_in[cur++]; p.gamma = (const float*)d_in[cur++]; }
        p.mean = (const float*)d_in[cur++];
        p.var  = (const float*)d_in[cur++];
      } else {
        p.gamma = p.beta = p.mean = p.var = nullptr;
      }
      p.Cin = dims[i]; p.Cout = dims[i + 1]; p.relu = (i < 2);
    }
  };
  if (posFirst) { fill(0, sa1d, true); fill(3, sa2d, true); fill(6, sa3d, true); fill(9, hdd, false); }
  else          { fill(9, hdd, false); fill(0, sa1d, true); fill(3, sa2d, true); fill(6, sa3d, true); }

  // --- workspace carving ---
  char* ws = (char*)d_ws;
  size_t off = 0;
  auto carve = [&](size_t bytes) -> char* {
    off = (off + 255) & ~(size_t)255;
    char* p = ws + off;
    off += bytes;
    return p;
  };

  h16* wt[12]; float* sc[12]; float* sh[12];
  for (int l = 0; l < 12; l++) {
    wt[l] = (h16*)carve((size_t)L[l].Cout * kpads[l] * sizeof(h16));
    sc[l] = (float*)carve((size_t)L[l].Cout * sizeof(float));
    sh[l] = (float*)carve((size_t)L[l].Cout * sizeof(float));
  }
  int*   idx1 = (int*)carve((size_t)B * S1 * sizeof(int));
  float* ctr1 = (float*)carve((size_t)B * S1 * 3 * sizeof(float));
  int*   nbr1 = (int*)carve((size_t)B * S1 * 64 * sizeof(int));
  h16*   x1   = (h16*)carve((size_t)B * S1 * 128 * sizeof(h16));
  int*   idx2 = (int*)carve((size_t)B * S2 * sizeof(int));
  float* ctr2 = (float*)carve((size_t)B * S2 * 3 * sizeof(float));
  int*   nbr2 = (int*)carve((size_t)B * S2 * 64 * sizeof(int));
  h16*   x2   = (h16*)carve((size_t)B * S2 * 256 * sizeof(h16));
  h16*   A3   = (h16*)carve((size_t)B * S2 * 288 * sizeof(h16));
  h16*   t31  = (h16*)carve((size_t)B * S2 * 256 * sizeof(h16));
  h16*   t32  = (h16*)carve((size_t)B * S2 * 512 * sizeof(h16));
  float* t33  = (float*)carve((size_t)B * S2 * 1024 * sizeof(float));
  float* gbuf = (float*)carve((size_t)B * 1024 * sizeof(float));
  (void)ws_size;

  // --- 1) pack weights (fold BN) ---
  for (int l = 0; l < 12; l++) {
    int total = L[l].Cout * kpads[l];
    pack_layer<<<(total + 255) / 256, 256, 0, stream>>>(
        L[l].W, L[l].b, L[l].gamma, L[l].beta, L[l].mean, L[l].var,
        wt[l], sc[l], sh[l], L[l].Cin, L[l].Cout, kpads[l]);
  }

  // --- 2) SA1: FPS -> radius -> fused edge-MLP + max ---
  size_t fshm1 = ((size_t)N * 3 + N + 512) * 4 + 512 * 4 + 4 * 4;
  fps_kernel<<<B, 512, fshm1, stream>>>(pos, idx1, ctr1, N, S1);
  ball_query<<<(B * S1 + 255) / 256, 256, 0, stream>>>(pos, ctr1, idx1, nbr1,
                                                       0.04f, N, S1, B);
  size_t shm1 = (size_t)(128 * 32 + 128 * 64 + 128 * 64 +
                         64 * 32 + 64 * 64 + 128 * 64) * 2 + 8 * 128 * 4;
  sa_fused<32, 1, 64, 64, 128, false><<<dim3(B, S1 / 2), 256, shm1, stream>>>(
      pos, ctr1, nbr1, nullptr,
      wt[0], sc[0], sh[0], wt[1], sc[1], sh[1], wt[2], sc[2], sh[2],
      x1, N, S1);

  // --- 3) SA2 ---
  size_t fshm2 = ((size_t)S1 * 3 + S1 + 512) * 4 + 512 * 4 + 4 * 4;
  fps_kernel<<<B, 512, fshm2, stream>>>(ctr1, idx2, ctr2, S1, S2);
  ball_query<<<(B * S2 + 255) / 256, 256, 0, stream>>>(ctr1, ctr2, idx2, nbr2,
                                                       0.16f, S1, S2, B);
  size_t shm2 = (size_t)(128 * 160 + 128 * 128 + 128 * 128 +
                         128 * 160 + 128 * 128 + 256 * 128) * 2 + 8 * 256 * 4;
  sa_fused<160, 128, 128, 128, 256, true><<<dim3(B, S2 / 2), 256, shm2, stream>>>(
      ctr1, ctr2, nbr2, x1,
      wt[3], sc[3], sh[3], wt[4], sc[4], sh[4], wt[5], sc[5], sh[5],
      x2, S1, S2);

  // --- 4) SA3 global MLP + max pool ---
  int M3 = B * S2;  // 4096
  build_a3<<<(M3 * 288 + 255) / 256, 256, 0, stream>>>(x2, ctr2, A3, M3);
  gemm_wmma<<<dim3(M3 / 128, 256 / 64), 256, 0, stream>>>(A3, 288, wt[6], sc[6], sh[6],
                                                          1, t31, nullptr, M3, 256);
  gemm_wmma<<<dim3(M3 / 128, 512 / 64), 256, 0, stream>>>(t31, 256, wt[7], sc[7], sh[7],
                                                          1, t32, nullptr, M3, 512);
  gemm_wmma<<<dim3(M3 / 128, 1024 / 64), 256, 0, stream>>>(t32, 512, wt[8], sc[8], sh[8],
                                                           0, nullptr, t33, M3, 1024);
  maxpool<<<(B * 1024 + 255) / 256, 256, 0, stream>>>(t33, gbuf, S2, 1024);

  // --- 5) head + log_softmax ---
  head_kernel<<<B, 256, 0, stream>>>(gbuf, wt[9], sh[9], wt[10], sh[10],
                                     wt[11], sh[11], (float*)d_out);
}
